// TinyRNNPolicy_51049981280507
// MI455X (gfx1250) — matrix operand order repaired and checked by reading, not compiled, not run on hardware
//
#include <hip/hip_runtime.h>

typedef __attribute__((ext_vector_type(16))) __bf16 v16bf;
typedef __attribute__((ext_vector_type(8)))  float  v8f;

#define N_NEUR   4096
#define NBLK     128          // persistent blocks (grid barrier participants)
#define NTHR     256          // 8 waves of 32
#define W_ELEMS  (16777216ULL)
// d_ws layout (bytes)
#define OFF_WSW  0ULL                       // bf16 swizzled W: 32 MiB
#define OFF_HA   33554432ULL                // bf16 h buffer A: 8 KiB
#define OFF_HB   (OFF_HA + 8192ULL)         // bf16 h buffer B: 8 KiB
#define OFF_PACT (OFF_HB + 8192ULL)         // float partial action sums: 2*64
#define OFF_BAR  (OFF_PACT + 512ULL)        // barrier: uint[64]
#define WS_NEED  (OFF_BAR + 256ULL)

__device__ __forceinline__ unsigned short f2bf(float f) {
    unsigned u = __float_as_uint(f);
    unsigned r = (u + 0x7FFFu + ((u >> 16) & 1u)) >> 16;   // RNE
    return (unsigned short)r;
}

// ---------------------------------------------------------------------------
// Convert fp32 row-major W -> bf16, swizzled into WMMA A-fragment-major tiles.
// Tile = (row-group g of 16 rows) x (k-chunk c of 32 cols) = 512 bf16 = 1 KiB.
// Within a tile, lane l (l<16)  holds row g*16+l, K = c*32 + {0..15};
//               lane l (l>=16) holds row g*16+l-16, K = c*32 + {16..31};
// stored contiguously: tile*512 + lane*16 + half*8   (elements).
// ---------------------------------------------------------------------------
__global__ void convert_w_kernel(const float* __restrict__ W,
                                 unsigned short* __restrict__ Wsw) {
    int c8 = blockIdx.x * blockDim.x + threadIdx.x;   // 0 .. 2^21-1 (8-elt chunks)
    int tile = c8 >> 6;          // 64 chunks per tile
    int q    = c8 & 63;
    int lane = q >> 1;
    int hh   = q & 1;
    int g  = tile >> 7;          // row group 0..255
    int cc = tile & 127;         // k-chunk  0..127
    int row = g * 16 + (lane & 15);
    int kb  = cc * 32 + (lane >> 4) * 16 + hh * 8;
    const float* src = W + (size_t)row * N_NEUR + kb;
    unsigned short* dst = Wsw + (size_t)tile * 512 + lane * 16 + hh * 8;
#pragma unroll
    for (int e = 0; e < 8; ++e) dst[e] = f2bf(src[e]);
}

__global__ void init_kernel(const float* __restrict__ h0,
                            unsigned short* __restrict__ hA,
                            unsigned* __restrict__ bar) {
    int t = blockIdx.x * blockDim.x + threadIdx.x;
    if (t < N_NEUR) hA[t] = f2bf(h0[t]);
    if (t == 0) { bar[0] = 0u; bar[32] = 0u; }
}

// ---------------------------------------------------------------------------
// Persistent RNN kernel: 128 blocks x 256 threads (8 waves).
// Block b owns rows [32b, 32b+32). Wave w: row-group rg=w>>2 (16 rows),
// K-slice ks=w&3 (1024 of K). One grid barrier per step.
// ---------------------------------------------------------------------------
__global__ void __launch_bounds__(NTHR)
rnn_kernel(const unsigned short* __restrict__ Wsw,
           unsigned short* __restrict__ hA,
           unsigned short* __restrict__ hB,
           float* __restrict__ pact,
           unsigned* __restrict__ bar,
           float* __restrict__ out,
           const int* __restrict__ nsp) {
    const int b    = blockIdx.x;
    const int tid  = threadIdx.x;
    const int wave = tid >> 5;
    const int lane = tid & 31;
    const int rg   = wave >> 2;        // 0..1 row-group within block
    const int ks   = wave & 3;         // 0..3 K-slice
    const int g    = 2 * b + rg;       // global row group 0..255
    const int n_steps = *nsp;

    // A fragments for this wave: tiles c = ks*32 .. ks*32+31 of group g
    const unsigned short* tbase =
        Wsw + (size_t)g * (128ULL * 512ULL) + (size_t)ks * 32ULL * 512ULL;
    const int lhi = (lane >> 4);       // 0 for lanes 0-15, 1 for 16-31

    __shared__ float part[8][16];
    unsigned my_gen = 0;

    unsigned short* hbuf0 = hA;
    unsigned short* hbuf1 = hB;

    for (int step = 0; step < n_steps; ++step) {
        const int par = step & 1;
        const unsigned short* hin  = par ? hbuf1 : hbuf0;
        unsigned short*       hout = par ? hbuf0 : hbuf1;

        // B base: lanes 0-15 need h[k0..k0+15], lanes 16-31 need h[k0+16..k0+31]
        const unsigned short* hb = hin + ks * 1024 + (lhi << 4);

        v8f c0 = {}; v8f c1 = {};
#pragma unroll 4
        for (int j = 0; j < 32; j += 2) {
            v16bf a0 = *(const v16bf*)(tbase + (size_t)j * 512 + lane * 16);
            v16bf b0 = *(const v16bf*)(hb + j * 32);
            v16bf a1 = *(const v16bf*)(tbase + (size_t)(j + 1) * 512 + lane * 16);
            v16bf b1 = *(const v16bf*)(hb + (j + 1) * 32);
            c0 = __builtin_amdgcn_wmma_f32_16x16x32_bf16(
                     false, a0, false, b0, (short)0, c0, false, false);
            c1 = __builtin_amdgcn_wmma_f32_16x16x32_bf16(
                     false, a1, false, b1, (short)0, c1, false, false);
        }
        v8f cs = c0 + c1;

        // Column 0 of D: lane 0 holds y[M=0..7] in VGPRs 0..7, lane 16 holds y[8..15]
        if (lane == 0) {
#pragma unroll
            for (int m = 0; m < 8; ++m) part[wave][m] = cs[m];
        } else if (lane == 16) {
#pragma unroll
            for (int m = 0; m < 8; ++m) part[wave][8 + m] = cs[m];
        }
        __syncthreads();

        // Reduce 4 K-slices, tanh, write bf16 h, per-block action partial.
        if (tid < 32) {
            const int rg2 = tid >> 4;
            const int m   = tid & 15;
            const int w0  = rg2 * 4;
            float s = part[w0][m] + part[w0 + 1][m] + part[w0 + 2][m] + part[w0 + 3][m];
            float y = tanhf(s);
            hout[b * 32 + tid] = f2bf(y);
            if (b < (NBLK / 2)) {        // rows < 2048 contribute to action mean
                float v = y;
#pragma unroll
                for (int off = 16; off > 0; off >>= 1)
                    v += __shfl_down(v, off, 32);
                if (tid == 0) pact[par * 64 + b] = v;
            }
        }
        __syncthreads();

        // ---- grid-wide barrier (release on arrive, acquire on depart) ----
        if (tid == 0) {
            unsigned old = __hip_atomic_fetch_add(&bar[0], 1u, __ATOMIC_ACQ_REL,
                                                  __HIP_MEMORY_SCOPE_AGENT);
            ++my_gen;
            if (old == (unsigned)(gridDim.x - 1)) {
                __hip_atomic_store(&bar[0], 0u, __ATOMIC_RELAXED,
                                   __HIP_MEMORY_SCOPE_AGENT);
                __hip_atomic_store(&bar[32], my_gen, __ATOMIC_RELEASE,
                                   __HIP_MEMORY_SCOPE_AGENT);
            } else {
                while (__hip_atomic_load(&bar[32], __ATOMIC_ACQUIRE,
                                         __HIP_MEMORY_SCOPE_AGENT) < my_gen)
                    __builtin_amdgcn_s_sleep(1);
            }
        }
        __syncthreads();

        // Block 0 folds the 64 partials in a fixed order (deterministic) and
        // emits the action; other blocks race ahead using the other parity.
        if (b == 0 && tid == 0) {
            const volatile float* pp = pact + par * 64;
            float tot = 0.f;
#pragma unroll 4
            for (int i = 0; i < 64; ++i) tot += pp[i];
            out[step] = tanhf(tot * (1.0f / 2048.0f));
        }
    }
}

extern "C" void kernel_launch(void* const* d_in, const int* in_sizes, int n_in,
                              void* d_out, int out_size, void* d_ws, size_t ws_size,
                              hipStream_t stream) {
    (void)in_sizes; (void)n_in; (void)out_size;
    if (ws_size < WS_NEED) return;

    const float* W  = (const float*)d_in[0];
    const float* h0 = (const float*)d_in[1];
    const int*  nsp = (const int*)d_in[2];
    float* out = (float*)d_out;

    char* ws = (char*)d_ws;
    unsigned short* Wsw = (unsigned short*)(ws + OFF_WSW);
    unsigned short* hA  = (unsigned short*)(ws + OFF_HA);
    unsigned short* hB  = (unsigned short*)(ws + OFF_HB);
    float*          pa  = (float*)(ws + OFF_PACT);
    unsigned*       bar = (unsigned*)(ws + OFF_BAR);

    // 1) fp32 -> bf16 fragment-swizzled W (one pass; 16.7M elts / 8 per thread)
    hipLaunchKernelGGL(convert_w_kernel, dim3(8192), dim3(256), 0, stream, W, Wsw);
    // 2) h0 -> bf16, reset barrier state
    hipLaunchKernelGGL(init_kernel, dim3(16), dim3(256), 0, stream, h0, hA, bar);
    // 3) persistent stepped RNN
    hipLaunchKernelGGL(rnn_kernel, dim3(NBLK), dim3(NTHR), 0, stream,
                       Wsw, hA, hB, pa, bar, out, nsp);
}
